// AAEncoder_2817498546719
// MI455X (gfx1250) — matrix operand order repaired and checked by reading, not compiled, not run on hardware
//
#include <hip/hip_runtime.h>

// ---------------------------------------------------------------------------
// Graph-attention encoder for MI455X (gfx1250, wave32, WMMA).
//
// Phase A (per node, VALU): rot/dpos geometry, center = LN(SIE(node_dxy)),
//   q' = center@wq^T / sqrt(8), hself/hgate gate precomputations, and the
//   j-only branch of _mie folded into c2[j].  ~80 MFLOP total -> trivial.
// Phase B (per node, WMMA): the O(N^2) neighbor pipeline.  One workgroup per
//   node n, 4 waves x 16-j tiles, chain of f16 WMMAs (f32 accumulate):
//     tmp1 -> @w2a(^T) + c2 -> LN/relu -> @aw(^T) -> LN -> @wk(^T)/@wv(^T)
//     scores via WMMA against block-diagonal Q', online softmax accumulate.
// Phase C (per node, VALU): @wo, gated residual, LN, 64->256->64 FFN.
//
// Input indices assume jax.tree_util dict flattening (keys sorted per level):
//  0 bos_mask  1 padding_mask  2 b_n1  3 b_n2  4 bhh  5 bih  6 bm1  7 bm2
//  8 bself  | center: 9 b1 10 b2 11 b3 12 be1 13 be2 14 be3 15 g1 16 g2 17 g3
//  18 w1 19 w2 20 w3 | 21 g_n1 22 g_n2 | nbr: 23 ab1 24 ab2 25 ag1 26 ag2
//  27 aw 28 awb | br0: 29 b1 30 b2 31 be1 32 g1 33 w1 34 w2 | br1: 35 b1
//  36 b2 37 be1 38 g1 39 w1 40 w2 | 41 whh 42 wih 43 wk 44 wm1 45 wm2 46 wo
//  47 wq 48 wself 49 wv | 50 positions 51 t
// ---------------------------------------------------------------------------

typedef __attribute__((ext_vector_type(16))) _Float16 v16h;
typedef __attribute__((ext_vector_type(8)))  _Float16 v8h;
typedef __attribute__((ext_vector_type(8)))  float    v8f;

#define TSTEPS   50
#define LN_EPS   1e-5f
#define NEGBIG   (-1e30f)

// ---------------- common helpers ----------------

__device__ __forceinline__ v8f wmma_f16(v16h a, v16h b, v8f c) {
  // D = A(16x32 f16) x B(32x16 f16) + C(16x16 f32)
  return __builtin_amdgcn_wmma_f32_16x16x32_f16(false, a, false, b, (short)0, c,
                                                false, false);
}

__device__ __forceinline__ v8f v8f_zero() {
  v8f z;
#pragma unroll
  for (int i = 0; i < 8; ++i) z[i] = 0.f;
  return z;
}

__device__ __forceinline__ float dot64(const float* w, const float* x) {
  float acc = 0.f;
#pragma unroll
  for (int k = 0; k < 64; ++k) acc = fmaf(w[k], x[k], acc);
  return acc;
}

// block of 64 threads
__device__ __forceinline__ float blk_sum64(float v, float* red) {
  int f = threadIdx.x;
  red[f] = v;
  __syncthreads();
#pragma unroll
  for (int off = 32; off > 0; off >>= 1) {
    if (f < off) red[f] += red[f + off];
    __syncthreads();
  }
  float s = red[0];
  __syncthreads();
  return s;
}

__device__ __forceinline__ float ln_act64(float y, const float* g, const float* be,
                                          bool do_relu, float* red) {
  float s  = blk_sum64(y, red);
  float s2 = blk_sum64(y * y, red);
  float mu  = s * (1.f / 64.f);
  float var = s2 * (1.f / 64.f) - mu * mu;
  float z = (y - mu) * rsqrtf(var + LN_EPS) * g[threadIdx.x] + be[threadIdx.x];
  return do_relu ? fmaxf(z, 0.f) : z;
}

// B fragment (32x16 f16) from f16 weight W stored row-major [n][k] in LDS.
// B(k,n) = W[n][k]; lanes 0-15 hold K=0..15, lanes 16-31 hold K=16..31 of
// the K-half; element e -> K = kbase + e (contiguous 32B run).
__device__ __forceinline__ v16h loadB16(const _Float16* W, int lane, int nt, int kh) {
  int col = lane & 15;
  int kb  = ((lane >> 4) << 4) + (kh << 5);
  return *(const v16h*)(W + (nt * 16 + col) * 64 + kb);
}

// A fragments (two 16x32 f16 covering K=0..63) from f16 tile S[16][64] in LDS.
// lane row = lane&15; element e -> K = kb + e, kb = 0 (lanes 0-15) or 16
// (lanes 16-31); ahi adds 32.
__device__ __forceinline__ void loadA16(const _Float16* S, int lane,
                                        v16h& alo, v16h& ahi) {
  int rr = lane & 15;
  int kb = (lane >> 4) << 4;
  const _Float16* p = S + rr * 64;
  alo = *(const v16h*)(p + kb);
  ahi = *(const v16h*)(p + kb + 32);
}

// ---------------- Phase A ----------------

__global__ __launch_bounds__(64) void phaseA(
    const float* __restrict__ positions, const int* __restrict__ tptr,
    const unsigned char* __restrict__ padmask,
    const float* cw1, const float* cb1, const float* cg1, const float* cbe1,
    const float* cw2, const float* cb2, const float* cg2, const float* cbe2,
    const float* cw3, const float* cb3, const float* cg3, const float* cbe3,
    const float* g_n1, const float* b_n1,
    const float* wq,
    const float* wself, const float* bself,
    const float* whh, const float* bhh,
    const float* b1w1, const float* b1b1, const float* b1g1, const float* b1be1,
    const float* b1w2, const float* b1b2,
    const float* b0b2,
    float* c2, float* qv, float* hself, float* hgate, float* posrot, float* padf) {
  __shared__ float red[64];
  __shared__ float act[64];
  __shared__ float tmpv[64];
  int n = blockIdx.x, f = threadIdx.x;
  int t = *tptr;

  float px = positions[((size_t)n * TSTEPS + t) * 2 + 0];
  float py = positions[((size_t)n * TSTEPS + t) * 2 + 1];
  float dx = px - positions[((size_t)n * TSTEPS + t - 1) * 2 + 0];
  float dy = py - positions[((size_t)n * TSTEPS + t - 1) * 2 + 1];
  float r = sqrtf(dx * dx + dy * dy);
  float rc, rs;
  if (r > 0.f) { rc = dx / r; rs = dy / r; } else { rc = 1.f; rs = 0.f; }
  float ndx = rc * dx - rs * dy;
  float ndy = rs * dx + rc * dy;

  // SIE(node_dxy) + LN(g_n1,b_n1) -> center
  float y = cw1[2 * f] * ndx + cw1[2 * f + 1] * ndy + cb1[f];
  float z = ln_act64(y, cg1, cbe1, true, red);
  act[f] = z; __syncthreads();
  y = dot64(cw2 + f * 64, act) + cb2[f];
  z = ln_act64(y, cg2, cbe2, true, red);
  act[f] = z; __syncthreads();
  y = dot64(cw3 + f * 64, act) + cb3[f];
  z = ln_act64(y, cg3, cbe3, false, red);
  z = ln_act64(z, g_n1, b_n1, false, red);      // center
  act[f] = z; __syncthreads();

  qv[(size_t)n * 64 + f]    = dot64(wq + f * 64, act) * 0.35355339059327f; // /sqrt(8)
  hself[(size_t)n * 64 + f] = dot64(wself + f * 64, act) + bself[f];
  hgate[(size_t)n * 64 + f] = dot64(whh + f * 64, act) + bhh[f];

  // c2[n] = branch1(dpos_n) + b2(branch0)
  y = b1w1[2 * f] * dx + b1w1[2 * f + 1] * dy + b1b1[f];
  z = ln_act64(y, b1g1, b1be1, true, red);
  tmpv[f] = z; __syncthreads();
  c2[(size_t)n * 64 + f] = dot64(b1w2 + f * 64, tmpv) + b1b2[f] + b0b2[f];

  if (f == 0) {
    posrot[n * 4 + 0] = px; posrot[n * 4 + 1] = py;
    posrot[n * 4 + 2] = rc; posrot[n * 4 + 3] = rs;
    padf[n] = padmask[(size_t)n * TSTEPS + t] ? 1.f : 0.f;
  }
}

// ---------------- Phase B (WMMA) ----------------

__global__ __launch_bounds__(128, 1) void phaseB(
    const float* __restrict__ posrot, const float* __restrict__ padf,
    const float* __restrict__ c2, const float* __restrict__ qv,
    const float* __restrict__ b0w1, const float* __restrict__ b0b1,
    const float* __restrict__ b0g1, const float* __restrict__ b0be1,
    const float* __restrict__ b0w2,
    const float* __restrict__ ag1, const float* __restrict__ ab1,
    const float* __restrict__ aw,  const float* __restrict__ awb,
    const float* __restrict__ ag2, const float* __restrict__ ab2,
    const float* __restrict__ wkm, const float* __restrict__ wvm,
    float* __restrict__ attnout, int N) {
  __shared__ __align__(32) _Float16 wlds[4][64 * 64];  // w2a, aw, wk, wv (f16)
  __shared__ float w1a_l[128];
  __shared__ float cvec[8][64];  // 0:b1 1:g1 2:be1 3:ag1 4:ab1 5:awb 6:ag2 7:ab2
  __shared__ __align__(32) _Float16 stage[4][16][64];  // per-wave f16 staging
  __shared__ float vstg[4][16 * 64];
  __shared__ float scstg[4][16 * 8];
  __shared__ int   mstg[4][16];
  __shared__ float mrg[4][32][4];

  int tid = threadIdx.x, wave = tid >> 5, lane = tid & 31;
  int n = blockIdx.x;

  for (int i = tid; i < 64 * 64; i += 128) {
    wlds[0][i] = (_Float16)b0w2[i];
    wlds[1][i] = (_Float16)aw[i];
    wlds[2][i] = (_Float16)wkm[i];
    wlds[3][i] = (_Float16)wvm[i];
  }
  w1a_l[tid] = b0w1[tid];  // 128 threads, 128 elems
  if (tid < 64) {
    cvec[0][tid] = b0b1[tid]; cvec[1][tid] = b0g1[tid]; cvec[2][tid] = b0be1[tid];
    cvec[3][tid] = ag1[tid];  cvec[4][tid] = ab1[tid];  cvec[5][tid] = awb[tid];
    cvec[6][tid] = ag2[tid];  cvec[7][tid] = ab2[tid];
  }
  __syncthreads();

  float pnx = posrot[n * 4 + 0], pny = posrot[n * 4 + 1];
  float rc = posrot[n * 4 + 2], rsn = posrot[n * 4 + 3];

  int col   = lane & 15;
  int base8 = (lane >> 4) << 3;

  // Q' B-fragments: Q'[k][h] = q[k]/sqrt(dh) iff k>>3==h (block-diagonal)
  v16h qb0, qb1;
#pragma unroll
  for (int kh = 0; kh < 2; ++kh) {
    int kb = ((lane >> 4) << 4) + (kh << 5);
    v16h b;
#pragma unroll
    for (int e = 0; e < 16; ++e) {
      int k = kb + e;
      float val = (col == (k >> 3)) ? qv[(size_t)n * 64 + k] : 0.f;
      b[e] = (_Float16)val;
    }
    if (kh == 0) qb0 = b; else qb1 = b;
  }

  float m_run = NEGBIG, l_run = 0.f, acc0 = 0.f, acc1 = 0.f;
  int f0 = lane * 2, f1 = lane * 2 + 1, head = lane >> 2;

  int niter = N >> 6;
  for (int it = 0; it < niter; ++it) {
    int jbase = it * 64 + wave * 16;

    // geometry + mask for this tile's rows (row = lane&15)
    float rr0, rr1;
    {
      int jl = lane & 15;
      int j = jbase + jl;
      float rx = posrot[j * 4 + 0] - pnx, ry = posrot[j * 4 + 1] - pny;
      float d2 = rx * rx + ry * ry;
      int mok = ((d2 <= 2500.f) && (padf[j] == 0.f)) || (j == n);
      mstg[wave][jl] = mok;
      rr0 = rc * rx - rsn * ry;
      rr1 = rsn * rx + rc * ry;
    }

    // branch0 layer1 (2->64) + LN + relu, built directly in A-fragment layout
    v16h alo, ahi;
    {
      float ssum = 0.f, ssq = 0.f;
#pragma unroll
      for (int e = 0; e < 16; ++e) {
        int k = (base8 << 1) + e;
        float y0 = fmaf(w1a_l[2 * k], rr0, fmaf(w1a_l[2 * k + 1], rr1, cvec[0][k]));
        int k2 = 32 + k;
        float y1 = fmaf(w1a_l[2 * k2], rr0, fmaf(w1a_l[2 * k2 + 1], rr1, cvec[0][k2]));
        ssum += y0 + y1; ssq += y0 * y0 + y1 * y1;
      }
      ssum += __shfl_xor(ssum, 16);
      ssq  += __shfl_xor(ssq, 16);
      float mu = ssum * (1.f / 64.f);
      float var = ssq * (1.f / 64.f) - mu * mu;
      float rsi = rsqrtf(var + LN_EPS);
#pragma unroll
      for (int e = 0; e < 16; ++e) {
        int k = (base8 << 1) + e;
        float y0 = fmaf(w1a_l[2 * k], rr0, fmaf(w1a_l[2 * k + 1], rr1, cvec[0][k]));
        alo[e] = (_Float16)fmaxf((y0 - mu) * rsi * cvec[1][k] + cvec[2][k], 0.f);
        int k2 = 32 + k;
        float y1 = fmaf(w1a_l[2 * k2], rr0, fmaf(w1a_l[2 * k2 + 1], rr1, cvec[0][k2]));
        ahi[e] = (_Float16)fmaxf((y1 - mu) * rsi * cvec[1][k2] + cvec[2][k2], 0.f);
      }
    }

    // h = tmp1 @ w2a^T + c2[j]
    v8f hc[4];
#pragma unroll
    for (int nt = 0; nt < 4; ++nt) {
      v8f c = wmma_f16(alo, loadB16(wlds[0], lane, nt, 0), v8f_zero());
      c = wmma_f16(ahi, loadB16(wlds[0], lane, nt, 1), c);
#pragma unroll
      for (int e = 0; e < 8; ++e)
        c[e] += c2[(size_t)(jbase + base8 + e) * 64 + nt * 16 + col];
      hc[nt] = c;
    }

    // relu(LN(h, ag1, ab1)) -> stage (f16)
#pragma unroll
    for (int e = 0; e < 8; ++e) {
      float s  = hc[0][e] + hc[1][e] + hc[2][e] + hc[3][e];
      float s2 = hc[0][e] * hc[0][e] + hc[1][e] * hc[1][e] +
                 hc[2][e] * hc[2][e] + hc[3][e] * hc[3][e];
#pragma unroll
      for (int mk = 1; mk <= 8; mk <<= 1) { s += __shfl_xor(s, mk); s2 += __shfl_xor(s2, mk); }
      float mu = s * (1.f / 64.f), var = s2 * (1.f / 64.f) - mu * mu;
      float rsi = rsqrtf(var + LN_EPS);
#pragma unroll
      for (int nt = 0; nt < 4; ++nt) {
        int fc = nt * 16 + col;
        float z = fmaxf((hc[nt][e] - mu) * rsi * cvec[3][fc] + cvec[4][fc], 0.f);
        stage[wave][base8 + e][fc] = (_Float16)z;
      }
    }
    __syncthreads();
    loadA16(&stage[wave][0][0], lane, alo, ahi);

    // a_out = g @ aw^T + awb
#pragma unroll
    for (int nt = 0; nt < 4; ++nt) {
      v8f c = wmma_f16(alo, loadB16(wlds[1], lane, nt, 0), v8f_zero());
      c = wmma_f16(ahi, loadB16(wlds[1], lane, nt, 1), c);
#pragma unroll
      for (int e = 0; e < 8; ++e) c[e] += cvec[5][nt * 16 + col];
      hc[nt] = c;
    }
    __syncthreads();

    // nbr = LN(a_out, ag2, ab2) -> stage (f16)
#pragma unroll
    for (int e = 0; e < 8; ++e) {
      float s  = hc[0][e] + hc[1][e] + hc[2][e] + hc[3][e];
      float s2 = hc[0][e] * hc[0][e] + hc[1][e] * hc[1][e] +
                 hc[2][e] * hc[2][e] + hc[3][e] * hc[3][e];
#pragma unroll
      for (int mk = 1; mk <= 8; mk <<= 1) { s += __shfl_xor(s, mk); s2 += __shfl_xor(s2, mk); }
      float mu = s * (1.f / 64.f), var = s2 * (1.f / 64.f) - mu * mu;
      float rsi = rsqrtf(var + LN_EPS);
#pragma unroll
      for (int nt = 0; nt < 4; ++nt) {
        int fc = nt * 16 + col;
        stage[wave][base8 + e][fc] = (_Float16)((hc[nt][e] - mu) * rsi * cvec[6][fc] + cvec[7][fc]);
      }
    }
    __syncthreads();
    loadA16(&stage[wave][0][0], lane, alo, ahi);

    // k = nbr @ wk^T ; v = nbr @ wv^T
    v8f kc[4];
#pragma unroll
    for (int nt = 0; nt < 4; ++nt) {
      v8f c = wmma_f16(alo, loadB16(wlds[2], lane, nt, 0), v8f_zero());
      kc[nt] = wmma_f16(ahi, loadB16(wlds[2], lane, nt, 1), c);
    }
    __syncthreads();
#pragma unroll
    for (int nt = 0; nt < 4; ++nt) {
      v8f c = wmma_f16(alo, loadB16(wlds[3], lane, nt, 0), v8f_zero());
      c = wmma_f16(ahi, loadB16(wlds[3], lane, nt, 1), c);
#pragma unroll
      for (int e = 0; e < 8; ++e) {
        vstg[wave][(base8 + e) * 64 + nt * 16 + col] = c[e];         // v (f32)
        stage[wave][base8 + e][nt * 16 + col] = (_Float16)kc[nt][e]; // k (f16)
      }
    }
    __syncthreads();

    // scores = k_tile @ Q'  (rows=j, cols=head)
    loadA16(&stage[wave][0][0], lane, alo, ahi);
    v8f sc = wmma_f16(alo, qb0, v8f_zero());
    sc = wmma_f16(ahi, qb1, sc);
    if (col < 8) {
#pragma unroll
      for (int e = 0; e < 8; ++e) {
        int jl = base8 + e;
        scstg[wave][jl * 8 + col] = mstg[wave][jl] ? sc[e] : NEGBIG;
      }
    }
    __syncthreads();

    // online softmax accumulate (lane owns features f0,f1; head = f0>>3)
#pragma unroll 4
    for (int jl = 0; jl < 16; ++jl) {
      float s = scstg[wave][jl * 8 + head];
      if (s > -1e29f) {
        float mn = fmaxf(m_run, s);
        float corr = __expf(m_run - mn);
        float p = __expf(s - mn);
        l_run = l_run * corr + p;
        acc0 = acc0 * corr + p * vstg[wave][jl * 64 + f0];
        acc1 = acc1 * corr + p * vstg[wave][jl * 64 + f1];
        m_run = mn;
      }
    }
    __syncthreads();
  }

  // merge per-wave softmax partials
  mrg[wave][lane][0] = m_run; mrg[wave][lane][1] = l_run;
  mrg[wave][lane][2] = acc0;  mrg[wave][lane][3] = acc1;
  __syncthreads();
  if (tid < 32) {
    float M = NEGBIG;
#pragma unroll
    for (int w = 0; w < 4; ++w) M = fmaxf(M, mrg[w][tid][0]);
    float L = 0.f, A0 = 0.f, A1 = 0.f;
#pragma unroll
    for (int w = 0; w < 4; ++w) {
      float e = __expf(mrg[w][tid][0] - M);
      L  += mrg[w][tid][1] * e;
      A0 += mrg[w][tid][2] * e;
      A1 += mrg[w][tid][3] * e;
    }
    attnout[(size_t)n * 64 + 2 * tid]     = A0 / L;
    attnout[(size_t)n * 64 + 2 * tid + 1] = A1 / L;
  }
}

// ---------------- Phase C ----------------

__global__ __launch_bounds__(64) void phaseC(
    const float* __restrict__ attnout, const float* __restrict__ hselfg,
    const float* __restrict__ hgateg,
    const float* __restrict__ wo, const float* __restrict__ wih,
    const float* __restrict__ bih,
    const float* __restrict__ g_n2, const float* __restrict__ b_n2,
    const float* __restrict__ wm1, const float* __restrict__ bm1,
    const float* __restrict__ wm2, const float* __restrict__ bm2,
    float* __restrict__ out) {
  __shared__ float red[64], am[64], nm[64], hn[64], u[256];
  int n = blockIdx.x, f = threadIdx.x;
  am[f] = attnout[(size_t)n * 64 + f];
  __syncthreads();
  float nmv = dot64(wo + f * 64, am);   // new_msg
  nm[f] = nmv;
  __syncthreads();
  float gs = dot64(wih + f * 64, nm) + bih[f] + hgateg[(size_t)n * 64 + f];
  float gate = 1.f / (1.f + __expf(-gs));
  float hh = nmv + gate * (hselfg[(size_t)n * 64 + f] - nmv);
  float hz = ln_act64(hh, g_n2, b_n2, false, red);
  hn[f] = hz;
  __syncthreads();
#pragma unroll
  for (int rb = 0; rb < 4; ++rb) {
    int rr = rb * 64 + f;
    u[rr] = fmaxf(dot64(wm1 + (size_t)rr * 64, hn) + bm1[rr], 0.f);
  }
  __syncthreads();
  float ff = bm2[f];
#pragma unroll 4
  for (int k = 0; k < 256; ++k) ff = fmaf(wm2[(size_t)f * 256 + k], u[k], ff);
  out[(size_t)n * 64 + f] = hz + ff;
}

// ---------------- launcher ----------------

extern "C" void kernel_launch(void* const* d_in, const int* in_sizes, int n_in,
                              void* d_out, int out_size, void* d_ws, size_t ws_size,
                              hipStream_t stream) {
  const float* positions        = (const float*)d_in[50];
  const int*   tptr             = (const int*)d_in[51];
  const unsigned char* padmask  = (const unsigned char*)d_in[1];
  int N = in_sizes[50] / (TSTEPS * 2);

  float* ws = (float*)d_ws;
  float* c2      = ws; ws += (size_t)N * 64;
  float* qv      = ws; ws += (size_t)N * 64;
  float* hself   = ws; ws += (size_t)N * 64;
  float* hgate   = ws; ws += (size_t)N * 64;
  float* posrot  = ws; ws += (size_t)N * 4;
  float* padf    = ws; ws += (size_t)N;
  float* attnout = ws; ws += (size_t)N * 64;

  phaseA<<<N, 64, 0, stream>>>(positions, tptr, padmask,
      (const float*)d_in[18], (const float*)d_in[9],  (const float*)d_in[15], (const float*)d_in[12],
      (const float*)d_in[19], (const float*)d_in[10], (const float*)d_in[16], (const float*)d_in[13],
      (const float*)d_in[20], (const float*)d_in[11], (const float*)d_in[17], (const float*)d_in[14],
      (const float*)d_in[21], (const float*)d_in[2],
      (const float*)d_in[47],
      (const float*)d_in[48], (const float*)d_in[8],
      (const float*)d_in[41], (const float*)d_in[4],
      (const float*)d_in[39], (const float*)d_in[35], (const float*)d_in[38], (const float*)d_in[37],
      (const float*)d_in[40], (const float*)d_in[36],
      (const float*)d_in[30],
      c2, qv, hself, hgate, posrot, padf);

  phaseB<<<N, 128, 0, stream>>>(posrot, padf, c2, qv,
      (const float*)d_in[33], (const float*)d_in[29], (const float*)d_in[32], (const float*)d_in[31],
      (const float*)d_in[34],
      (const float*)d_in[25], (const float*)d_in[23],
      (const float*)d_in[27], (const float*)d_in[28],
      (const float*)d_in[26], (const float*)d_in[24],
      (const float*)d_in[43], (const float*)d_in[49],
      attnout, N);

  phaseC<<<N, 64, 0, stream>>>(attnout, hself, hgate,
      (const float*)d_in[46], (const float*)d_in[42], (const float*)d_in[5],
      (const float*)d_in[22], (const float*)d_in[3],
      (const float*)d_in[44], (const float*)d_in[6], (const float*)d_in[45], (const float*)d_in[7],
      (float*)d_out);
}
